// WorldStateEncoder_39874476376567
// MI455X (gfx1250) — matrix operand order repaired and checked by reading, not compiled, not run on hardware
//
#include <hip/hip_runtime.h>
#include <hip/hip_bf16.h>
#include <math.h>

typedef __attribute__((ext_vector_type(16))) __bf16 v16bf;
typedef __attribute__((ext_vector_type(8)))  __bf16 v8bf;   // 16B, align 16
typedef __attribute__((ext_vector_type(8)))  float  v8f;

#define DEVINL __device__ __forceinline__

DEVINL float lrelu02(float v) { return v > 0.f ? v : 0.2f * v; }
DEVINL float elu1(float v)    { return v > 0.f ? v : (expf(v) - 1.f); }

// Monotonic order-preserving float<->uint mapping for atomicMax-based segment max.
DEVINL unsigned fkey(float x) {
    unsigned u = __float_as_uint(x);
    return (u & 0x80000000u) ? ~u : (u | 0x80000000u);
}
DEVINL float funkey(unsigned k) {
    unsigned u = (k & 0x80000000u) ? (k & 0x7FFFFFFFu) : ~k;
    return __uint_as_float(u);
}
// fkey(-inf) = 0x007FFFFF : init value for the max buffers.
#define NEG_INF_KEY 0x007FFFFFu

DEVINL void edge_sd(const int* __restrict__ ei, int E, int e, int& s, int& d) {
    if (e < E) { s = ei[e]; d = ei[E + e]; }
    else       { s = d = e - E; }          // appended self loops
}

// ---------------------------------------------------------------------------
// WMMA GEMM: out[M,Nc] = act(A[M,K] @ W[K,Nc] + bias).  fp32 in/out, bf16 MACs.
// One wave -> 16x128 strip (8 accumulators share one A fragment).
// Block = 8 waves -> 128x128 tile.  Requires M%128==0, K%32==0, Nc%128==0.
// B tile double-buffered through LDS (bf16, [n][k]); rows padded to 40 elems:
//  * fragment base nn*80 + 32*half bytes -> 16B aligned -> ds_load_b128
//  * bank stride 20 dwords; {20*nn mod 64 : nn=0..15} 4-dword groups tile all
//    64 banks exactly once -> conflict-free
//  * fragment immediate offsets (t*1280B <= 8960) fit the 16-bit DS offset
// ---------------------------------------------------------------------------
#define LB_PAD 40

__global__ __launch_bounds__(256)
void gemm_bias_act_wmma(const float* __restrict__ A, const float* __restrict__ Wm,
                        const float* __restrict__ bias, float* __restrict__ out,
                        int M, int K, int Nc, int act)
{
    __shared__ __bf16 lB[2][128 * LB_PAD];
    const int lane = threadIdx.x & 31;
    const int wave = threadIdx.x >> 5;
    const int m0 = (blockIdx.x * 8 + wave) * 16;
    const int n0 = blockIdx.y * 128;
    const int nn = lane & 15;               // row (A) / col (B/C) within tile
    const int half = lane >> 4;

    // Stage W[kk..kk+32)[n0..n0+128) -> bf16 LDS tile (transposed to [n][k]).
    auto stage = [&](int buf, int kk) {
        #pragma unroll
        for (int r = 0; r < 16; ++r) {
            int idx = threadIdx.x + r * 256;       // 4096 elements
            int k = idx >> 7, n = idx & 127;
            lB[buf][n * LB_PAD + k] = (__bf16)Wm[(size_t)(kk + k) * Nc + n0 + n];
        }
    };

    // 16B-aligned B-fragment load: two ds_load_b128.
    auto loadB = [&](const __bf16* lb) {
        v8bf lo = *reinterpret_cast<const v8bf*>(lb);
        v8bf hi = *reinterpret_cast<const v8bf*>(lb + 8);
        v16bf b;
        #pragma unroll
        for (int j = 0; j < 8; ++j) { b[j] = lo[j]; b[8 + j] = hi[j]; }
        return b;
    };

    v8f acc[8] = {};
    stage(0, 0);

    for (int kk = 0; kk < K; kk += 32) {
        const int buf = (kk >> 5) & 1;
        __syncthreads();                           // staging of `buf` complete
        if (kk + 32 < K) {
            stage(buf ^ 1, kk + 32);               // overlap next tile with compute
            __builtin_prefetch(A + (size_t)(m0 + nn) * K + kk + 32, 0, 0);
        }

        // A fragment: VGPR j holds K = 2j + 16*half (contiguous, matches B).
        v16bf a;
        const float* Arow = A + (size_t)(m0 + nn) * K + kk + 16 * half;
        #pragma unroll
        for (int j = 0; j < 8; ++j) {
            int k0 = 2 * j;
            float2 f = *reinterpret_cast<const float2*>(Arow + k0);
            a[2 * j]     = (__bf16)f.x;
            a[2 * j + 1] = (__bf16)f.y;
        }

        // B fragments: lanes 0-15 hold K=0..15, lanes 16-31 hold K=16..31.
        // Software pipeline: load fragment t+1 while WMMA t is issued.
        const __bf16* lbase = &lB[buf][nn * LB_PAD + 16 * half];
        v16bf bcur = loadB(lbase);
        v16bf bnxt;
        #pragma unroll
        for (int t = 0; t < 8; ++t) {
            if (t < 7) bnxt = loadB(lbase + (t + 1) * 16 * LB_PAD);
            acc[t] = __builtin_amdgcn_wmma_f32_16x16x32_bf16(
                false, a, false, bcur, (short)0, acc[t], false, false);
            bcur = bnxt;
        }
    }

    if (m0 < M) {
        #pragma unroll
        for (int t = 0; t < 8; ++t) {
            #pragma unroll
            for (int vg = 0; vg < 8; ++vg) {
                int m = m0 + vg + 8 * half;     // C/D layout: VGPR r -> M=r (+8 for hi lanes)
                int n = n0 + t * 16 + nn;
                float v = acc[t][vg];
                if (bias) v += bias[n];
                if (act == 1)      v = fmaxf(v, 0.f);
                else if (act == 2) v = elu1(v);
                out[(size_t)m * Nc + n] = v;
            }
        }
    }
}

// ---------------------------------------------------------------------------
// Fill kernels (graph-capture safe replacement for memset)
// ---------------------------------------------------------------------------
__global__ void fill_f32(float* p, float v, int n) {
    int i = blockIdx.x * blockDim.x + threadIdx.x;
    if (i < n) p[i] = v;
}
__global__ void fill_u32(unsigned* p, unsigned v, int n) {
    int i = blockIdx.x * blockDim.x + threadIdx.x;
    if (i < n) p[i] = v;
}

// ---------------------------------------------------------------------------
// Attention coefficients: a_src[n,h] = <x[n,h,:], att_src[h,:]>, same for dst.
// One wave per (n,h) row; x stored [N, H*C] so row (n*4+h) starts at row*256.
// ---------------------------------------------------------------------------
__global__ __launch_bounds__(256)
void att_reduce(const float* __restrict__ x, const float* __restrict__ att_src,
                const float* __restrict__ att_dst,
                float* __restrict__ a_src, float* __restrict__ a_dst, int nrows)
{
    int row = blockIdx.x * 8 + (threadIdx.x >> 5);
    if (row >= nrows) return;
    int lane = threadIdx.x & 31;
    int h = row & 3;
    const float* xp = x + (size_t)row * 256;
    const float* as = att_src + h * 256;
    const float* ad = att_dst + h * 256;
    float s0 = 0.f, s1 = 0.f;
    for (int c = lane; c < 256; c += 32) {
        float xv = xp[c];
        s0 += xv * as[c];
        s1 += xv * ad[c];
    }
    #pragma unroll
    for (int off = 16; off; off >>= 1) {
        s0 += __shfl_down(s0, off, 32);
        s1 += __shfl_down(s1, off, 32);
    }
    if (lane == 0) { a_src[row] = s0; a_dst[row] = s1; }
}

// ---------------------------------------------------------------------------
// Edge softmax passes
// ---------------------------------------------------------------------------
__global__ void edge_max(const int* __restrict__ ei, const float* __restrict__ a_src,
                         const float* __restrict__ a_dst, unsigned* __restrict__ emax,
                         int E, int ET)
{
    int e = blockIdx.x * blockDim.x + threadIdx.x;
    if (e >= ET) return;
    int s, d; edge_sd(ei, E, e, s, d);
    #pragma unroll
    for (int h = 0; h < 4; ++h) {
        float v = lrelu02(a_src[s * 4 + h] + a_dst[d * 4 + h]);
        atomicMax(&emax[d * 4 + h], fkey(v));
    }
}

__global__ void edge_exp(const int* __restrict__ ei, const float* __restrict__ a_src,
                         const float* __restrict__ a_dst, const unsigned* __restrict__ emax,
                         float* __restrict__ p, float* __restrict__ denom, int E, int ET)
{
    int e = blockIdx.x * blockDim.x + threadIdx.x;
    if (e >= ET) return;
    int s, d; edge_sd(ei, E, e, s, d);
    #pragma unroll
    for (int h = 0; h < 4; ++h) {
        float v = lrelu02(a_src[s * 4 + h] + a_dst[d * 4 + h]);
        float mx = funkey(emax[d * 4 + h]);
        if (!isfinite(mx)) mx = 0.f;
        float pe = expf(v - mx);
        p[(size_t)e * 4 + h] = pe;
        atomicAdd(&denom[d * 4 + h], pe);
    }
}

__global__ void edge_norm(const int* __restrict__ ei, float* __restrict__ alpha,
                          const float* __restrict__ denom, int E, int ET)
{
    int idx = blockIdx.x * blockDim.x + threadIdx.x;
    if (idx >= ET * 4) return;
    int e = idx >> 2, h = idx & 3;
    int s, d; edge_sd(ei, E, e, s, d);
    alpha[idx] = alpha[idx] / (denom[d * 4 + h] + 1e-16f);
}

// Heavy pass: out[dst,h,c] += alpha[e,h] * x[src,h,c].  One thread per 4 channels.
__global__ void edge_aggregate(const int* __restrict__ ei, const float* __restrict__ x,
                               const float* __restrict__ alpha, float* __restrict__ out,
                               int E, int ET)
{
    long long idx = (long long)blockIdx.x * blockDim.x + threadIdx.x;
    long long tot = (long long)ET * 256;
    if (idx >= tot) return;
    int e = (int)(idx >> 8);
    int q = (int)(idx & 255);      // quad within 1024 channels
    int h = q >> 6;
    int s, d; edge_sd(ei, E, e, s, d);
    float al = alpha[(size_t)e * 4 + h];
    float4 xv = *reinterpret_cast<const float4*>(x + (size_t)s * 1024 + q * 4);
    float* op = out + (size_t)d * 1024 + q * 4;
    atomicAdd(op + 0, al * xv.x);
    atomicAdd(op + 1, al * xv.y);
    atomicAdd(op + 2, al * xv.z);
    atomicAdd(op + 3, al * xv.w);
}

// ---------------------------------------------------------------------------
// Post-aggregation finalize
// ---------------------------------------------------------------------------
__global__ void bias_elu(float* h, const float* __restrict__ b, int n) {
    int i = blockIdx.x * blockDim.x + threadIdx.x;
    if (i < n) h[i] = elu1(h[i] + b[i & 1023]);
}

__global__ void head_mean_elu(const float* __restrict__ agg, const float* __restrict__ b,
                              float* __restrict__ out, int N)
{
    int idx = blockIdx.x * blockDim.x + threadIdx.x;
    if (idx >= N * 256) return;
    int n = idx >> 8, c = idx & 255;
    const float* r = agg + (size_t)n * 1024;
    float s = (r[c] + r[256 + c] + r[512 + c] + r[768 + c]) * 0.25f + b[c];
    out[idx] = elu1(s);
}

// ---------------------------------------------------------------------------
// Global mean pool by batch id
// ---------------------------------------------------------------------------
__global__ void pool_sum(const float* __restrict__ h, const int* __restrict__ batch,
                         float* __restrict__ pooled, float* __restrict__ cnt, int N)
{
    int idx = blockIdx.x * blockDim.x + threadIdx.x;
    if (idx >= N * 256) return;
    int n = idx >> 8, c = idx & 255;
    int g = batch[n];
    atomicAdd(&pooled[g * 256 + c], h[idx]);
    if (c == 0) atomicAdd(&cnt[g], 1.f);
}
__global__ void pool_div(float* pooled, const float* cnt, int B) {
    int idx = blockIdx.x * blockDim.x + threadIdx.x;
    if (idx >= B * 256) return;
    pooled[idx] /= fmaxf(cnt[idx >> 8], 1.f);
}

// ---------------------------------------------------------------------------
// Small fp32 GEMM for B=8 rows (pooled proj / test MLP / fusion)
// ---------------------------------------------------------------------------
__global__ void small_gemm(const float* __restrict__ A, const float* __restrict__ W,
                           const float* __restrict__ bias, float* __restrict__ out,
                           int M, int K, int Nc, int act)
{
    int idx = blockIdx.x * blockDim.x + threadIdx.x;
    if (idx >= M * Nc) return;
    int m = idx / Nc, n = idx % Nc;
    float s = bias ? bias[n] : 0.f;
    const float* a = A + (size_t)m * K;
    for (int k = 0; k < K; ++k) s += a[k] * W[(size_t)k * Nc + n];
    if (act == 1) s = fmaxf(s, 0.f);
    out[idx] = s;
}

__global__ void concat_kernel(const float* m, const float* d, const float* mg,
                              const float* t, float* comb, int B)
{
    int idx = blockIdx.x * blockDim.x + threadIdx.x;
    if (idx >= B * 896) return;
    int r = idx / 896, j = idx % 896;
    float v;
    if (j < 256)      v = m[r * 256 + j];
    else if (j < 512) v = d[r * 256 + (j - 256)];
    else if (j < 768) v = mg[r * 256 + (j - 512)];
    else              v = t[r * 128 + (j - 768)];
    comb[idx] = v;
}

// LayerNorm(512) + relu, one 256-thread block per row.
__global__ __launch_bounds__(256)
void layernorm_relu(float* __restrict__ h, const float* __restrict__ g,
                    const float* __restrict__ b)
{
    __shared__ float sh[8], sh2[8];
    int row = blockIdx.x, t = threadIdx.x, lane = t & 31, wave = t >> 5;
    float x0 = h[row * 512 + t], x1 = h[row * 512 + 256 + t];
    float s = x0 + x1;
    #pragma unroll
    for (int off = 16; off; off >>= 1) s += __shfl_down(s, off, 32);
    if (lane == 0) sh[wave] = s;
    __syncthreads();
    if (t == 0) { float tot = 0; for (int i = 0; i < 8; ++i) tot += sh[i]; sh[0] = tot / 512.f; }
    __syncthreads();
    float mu = sh[0];
    float d0 = x0 - mu, d1 = x1 - mu;
    float v = d0 * d0 + d1 * d1;
    #pragma unroll
    for (int off = 16; off; off >>= 1) v += __shfl_down(v, off, 32);
    if (lane == 0) sh2[wave] = v;
    __syncthreads();
    if (t == 0) { float tot = 0; for (int i = 0; i < 8; ++i) tot += sh2[i]; sh2[0] = rsqrtf(tot / 512.f + 1e-5f); }
    __syncthreads();
    float rstd = sh2[0];
    h[row * 512 + t]       = fmaxf(d0 * rstd * g[t] + b[t], 0.f);
    h[row * 512 + 256 + t] = fmaxf(d1 * rstd * g[t + 256] + b[t + 256], 0.f);
}

// ---------------------------------------------------------------------------
// Host orchestration
// ---------------------------------------------------------------------------
static inline int cdiv(long long a, long long b) { return (int)((a + b - 1) / b); }

struct Enc {
    const float *emb_W, *emb_b;
    const float *W[3], *as_[3], *ad_[3], *b_[3];
    const float *out_W, *out_b;
};

static void run_encoder(const float* x_in, const int* ei, const int* batch, const Enc& P,
                        float* enc_out, float* buf_h, float* buf_x, float* a_src,
                        float* a_dst, unsigned* emax, float* denom, float* alpha,
                        float* pooled, float* cnt, int N, int E, int B, hipStream_t s)
{
    const int T = 256;
    const int ET = E + N;
    // embedding: relu(x @ emb_W + emb_b) -> buf_h [N,256]
    gemm_bias_act_wmma<<<dim3(N / 128, 256 / 128), T, 0, s>>>(
        x_in, P.emb_W, P.emb_b, buf_h, N, 128, 256, 1);
    for (int L = 0; L < 3; ++L) {
        int Kd = (L == 0) ? 256 : 1024;
        gemm_bias_act_wmma<<<dim3(N / 128, 1024 / 128), T, 0, s>>>(
            buf_h, P.W[L], nullptr, buf_x, N, Kd, 1024, 0);
        att_reduce<<<cdiv((long long)N * 4, 8), T, 0, s>>>(
            buf_x, P.as_[L], P.ad_[L], a_src, a_dst, N * 4);
        fill_u32<<<cdiv(N * 4, T), T, 0, s>>>(emax, NEG_INF_KEY, N * 4);
        fill_f32<<<cdiv(N * 4, T), T, 0, s>>>(denom, 0.f, N * 4);
        fill_f32<<<cdiv((long long)N * 1024, T), T, 0, s>>>(buf_h, 0.f, N * 1024);
        edge_max<<<cdiv(ET, T), T, 0, s>>>(ei, a_src, a_dst, emax, E, ET);
        edge_exp<<<cdiv(ET, T), T, 0, s>>>(ei, a_src, a_dst, emax, alpha, denom, E, ET);
        edge_norm<<<cdiv((long long)ET * 4, T), T, 0, s>>>(ei, alpha, denom, E, ET);
        edge_aggregate<<<cdiv((long long)ET * 256, T), T, 0, s>>>(ei, buf_x, alpha, buf_h, E, ET);
        if (L < 2)
            bias_elu<<<cdiv((long long)N * 1024, T), T, 0, s>>>(buf_h, P.b_[L], N * 1024);
        else
            head_mean_elu<<<cdiv((long long)N * 256, T), T, 0, s>>>(buf_h, P.b_[2], buf_x, N);
    }
    fill_f32<<<cdiv(B * 256, T), T, 0, s>>>(pooled, 0.f, B * 256);
    fill_f32<<<1, 32, 0, s>>>(cnt, 0.f, B);
    pool_sum<<<cdiv((long long)N * 256, T), T, 0, s>>>(buf_x, batch, pooled, cnt, N);
    pool_div<<<cdiv(B * 256, T), T, 0, s>>>(pooled, cnt, B);
    small_gemm<<<cdiv(B * 256, T), T, 0, s>>>(pooled, P.out_W, P.out_b, enc_out, B, 256, 256, 0);
}

extern "C" void kernel_launch(void* const* d_in, const int* in_sizes, int n_in,
                              void* d_out, int out_size, void* d_ws, size_t ws_size,
                              hipStream_t stream)
{
    const float* module_x       = (const float*)d_in[0];
    const int*   module_ei      = (const int*)d_in[1];
    const int*   module_batch   = (const int*)d_in[2];
    const float* dispatch_x     = (const float*)d_in[3];
    const int*   dispatch_ei    = (const int*)d_in[4];
    const int*   dispatch_batch = (const int*)d_in[5];
    const float* method_agg     = (const float*)d_in[6];
    const float* test_stats     = (const float*)d_in[7];
    const int N = in_sizes[0] / 128;
    const int E = in_sizes[1] / 2;
    const int B = in_sizes[6] / 256;

    // Params flattened in setup_inputs insertion order: mod(8..23), dis(24..39),
    // test(40..43), fusion(44..49).
    auto encp = [&](int base) {
        Enc P;
        P.emb_W = (const float*)d_in[base];
        P.emb_b = (const float*)d_in[base + 1];
        for (int i = 0; i < 3; ++i) {
            P.W[i]   = (const float*)d_in[base + 2 + 4 * i];
            P.as_[i] = (const float*)d_in[base + 3 + 4 * i];
            P.ad_[i] = (const float*)d_in[base + 4 + 4 * i];
            P.b_[i]  = (const float*)d_in[base + 5 + 4 * i];
        }
        P.out_W = (const float*)d_in[base + 14];
        P.out_b = (const float*)d_in[base + 15];
        return P;
    };
    Enc Pm = encp(8), Pd = encp(24);
    const float* tW1 = (const float*)d_in[40];
    const float* tb1 = (const float*)d_in[41];
    const float* tW2 = (const float*)d_in[42];
    const float* tb2 = (const float*)d_in[43];
    const float* fW1 = (const float*)d_in[44];
    const float* fb1 = (const float*)d_in[45];
    const float* lng = (const float*)d_in[46];
    const float* lnb = (const float*)d_in[47];
    const float* fW2 = (const float*)d_in[48];
    const float* fb2 = (const float*)d_in[49];

    // Workspace carve (bump allocator, 256B aligned)
    uintptr_t w = (uintptr_t)d_ws;
    auto alloc = [&](size_t bytes) {
        uintptr_t p = (w + 255) & ~(uintptr_t)255;
        w = p + bytes;
        return (void*)p;
    };
    float*    buf_h  = (float*)alloc((size_t)N * 1024 * 4);
    float*    buf_x  = (float*)alloc((size_t)N * 1024 * 4);
    float*    a_src  = (float*)alloc((size_t)N * 4 * 4);
    float*    a_dst  = (float*)alloc((size_t)N * 4 * 4);
    unsigned* emax   = (unsigned*)alloc((size_t)N * 4 * 4);
    float*    denom  = (float*)alloc((size_t)N * 4 * 4);
    float*    alpha  = (float*)alloc((size_t)(E + N) * 4 * 4);
    float*    pooled = (float*)alloc((size_t)B * 256 * 4);
    float*    cnt    = (float*)alloc((size_t)B * 4);
    float*    enc_m  = (float*)alloc((size_t)B * 256 * 4);
    float*    enc_d  = (float*)alloc((size_t)B * 256 * 4);
    float*    t1     = (float*)alloc((size_t)B * 128 * 4);
    float*    t2     = (float*)alloc((size_t)B * 128 * 4);
    float*    comb   = (float*)alloc((size_t)B * 896 * 4);
    float*    fh     = (float*)alloc((size_t)B * 512 * 4);
    (void)ws_size; (void)n_in; (void)out_size;

    run_encoder(module_x, module_ei, module_batch, Pm, enc_m,
                buf_h, buf_x, a_src, a_dst, emax, denom, alpha, pooled, cnt,
                N, E, B, stream);
    run_encoder(dispatch_x, dispatch_ei, dispatch_batch, Pd, enc_d,
                buf_h, buf_x, a_src, a_dst, emax, denom, alpha, pooled, cnt,
                N, E, B, stream);

    const int T = 256;
    small_gemm<<<cdiv(B * 128, T), T, 0, stream>>>(test_stats, tW1, tb1, t1, B, 4, 128, 1);
    small_gemm<<<cdiv(B * 128, T), T, 0, stream>>>(t1, tW2, tb2, t2, B, 128, 128, 0);
    concat_kernel<<<cdiv(B * 896, T), T, 0, stream>>>(enc_m, enc_d, method_agg, t2, comb, B);
    small_gemm<<<cdiv(B * 512, T), T, 0, stream>>>(comb, fW1, fb1, fh, B, 896, 512, 0);
    layernorm_relu<<<B, 256, 0, stream>>>(fh, lng, lnb);
    small_gemm<<<cdiv(B * 512, T), T, 0, stream>>>(fh, fW2, fb2, (float*)d_out, B, 512, 512, 0);
}
